// CEmbedder_L_15075335209341
// MI455X (gfx1250) — matrix operand order repaired and checked by reading, not compiled, not run on hardware
//
#include <hip/hip_runtime.h>

typedef _Float16 v16h __attribute__((ext_vector_type(16)));
typedef _Float16 v8h  __attribute__((ext_vector_type(8)));
typedef float    v8f  __attribute__((ext_vector_type(8)));
typedef float    v4f  __attribute__((ext_vector_type(4)));

#define B_TOT   2048
#define NBRANCH 36
#define HID     512
#define ED      1024

#define DBLK    128   // d (embed) columns per block
#define MBLK    256   // batch rows per block = 8 waves * 32
#define KSTAGE  64    // K chunk staged in LDS (2 WMMA K-steps)
#define NSTAGE  (HID / KSTAGE)   // 8
#define LDS_ROW 72    // halves per d-row: 64 + 8 pad (144 B -> conflict-friendly)

__global__ __launch_bounds__(256)
void fused_mlp_wmma_kernel(const float* __restrict__ x,
                           const float* __restrict__ W1,
                           const float* __restrict__ b1,
                           const float* __restrict__ W2,
                           const float* __restrict__ b2,
                           float* __restrict__ out)
{
    __shared__ __attribute__((aligned(16))) _Float16 lds[2][DBLK * LDS_ROW];

    const int i    = blockIdx.z;              // branch 0..35
    const int DB   = blockIdx.x * DBLK;       // embed-dim base
    const int tid  = threadIdx.x;
    const int lane = tid & 31;
    const int wave = tid >> 5;
    const int MB   = blockIdx.y * MBLK + wave * 32;  // batch base for this wave
    const int l15  = lane & 15;
    const int lhi  = lane >> 4;               // 0 or 1

    // ---- cooperative W2 staging mapping: thread -> (4 consecutive d, 8 consecutive k)
    const int s_d0  = (tid & 31) * 4;         // 0..124
    const int s_kb0 = (tid >> 5) * 8;         // 0..56

    const float* W2i = W2 + (size_t)i * HID * ED;
    const float* W1i = W1 + (size_t)i * HID;
    const float* b1i = b1 + (size_t)i * HID;

    v4f stg[8];

    auto issue_loads = [&](int s) {
        const float* src = W2i + (size_t)(s * KSTAGE + s_kb0) * ED + DB + s_d0;
#pragma unroll
        for (int u = 0; u < 8; ++u)                 // k = s*64 + s_kb0 + u
            stg[u] = *(const v4f*)(src + (size_t)u * ED);
    };
    auto cvt_store = [&](int buf) {
#pragma unroll
        for (int c = 0; c < 4; ++c) {               // transpose 4d x 8k in regs
            v8h row;
#pragma unroll
            for (int u = 0; u < 8; ++u)
                row[u] = (_Float16)stg[u][c];
            *(v8h*)&lds[buf][(s_d0 + c) * LDS_ROW + s_kb0] = row;
        }
    };

    // ---- per-wave scalar x values (A of the rank-1 fc1), one per m-tile row
    const float xm0 = x[(size_t)(MB +      l15) * NBRANCH + i];
    const float xm1 = x[(size_t)(MB + 16 + l15) * NBRANCH + i];

    // ---- accumulators initialized with bias b2 (C operand of first WMMA)
    v8f acc[2][8];
#pragma unroll
    for (int dt = 0; dt < 8; ++dt) {
        const float* bp = b2 + (size_t)i * ED + DB + dt * 16 + lhi * 8;
        v4f blo = *(const v4f*)bp;
        v4f bhi = *(const v4f*)(bp + 4);
        v8f c;
        c[0] = blo[0]; c[1] = blo[1]; c[2] = blo[2]; c[3] = blo[3];
        c[4] = bhi[0]; c[5] = bhi[1]; c[6] = bhi[2]; c[7] = bhi[3];
        acc[0][dt] = c;
        acc[1][dt] = c;
    }

    // ---- prologue: stage 0 into LDS buffer 0
    issue_loads(0);
    cvt_store(0);
    __syncthreads();

    for (int s = 0; s < NSTAGE; ++s) {
        const int buf = s & 1;
        if (s + 1 < NSTAGE) issue_loads(s + 1);     // prefetch next W2 chunk

#pragma unroll
        for (int ks = 0; ks < 2; ++ks) {            // two K=32 steps per stage
            // -------- build B fragments (h^T) in registers --------
            // B layout 32x16: lane -> col m=l15, k = lhi*16 + j (j=0..15 contiguous)
            const int kg = s * KSTAGE + ks * 32 + lhi * 16;
            float wk[16], bk[16];
#pragma unroll
            for (int q = 0; q < 4; ++q) {
                v4f wv = *(const v4f*)(W1i + kg + q * 4);
                v4f bv = *(const v4f*)(b1i + kg + q * 4);
#pragma unroll
                for (int c = 0; c < 4; ++c) { wk[q * 4 + c] = wv[c]; bk[q * 4 + c] = bv[c]; }
            }
            v16h hfrag[2];
#pragma unroll
            for (int mt = 0; mt < 2; ++mt) {
                const float xm = mt ? xm1 : xm0;
                v16h hf;
#pragma unroll
                for (int j = 0; j < 16; ++j) {
                    float h = fmaf(xm, wk[j], bk[j]);
                    h = fmaxf(h, 0.01f * h);        // LeakyReLU(0.01)
                    hf[j] = (_Float16)h;
                }
                hfrag[mt] = hf;
            }

            // -------- A fragments (W2^T) from LDS + WMMA --------
            // A layout 16x32: lane row M=d=l15; k = lhi*16 + j (j=0..15 contiguous)
            const _Float16* abase = &lds[buf][0] + ks * 32 + lhi * 16;
#pragma unroll
            for (int dt = 0; dt < 8; ++dt) {
                const _Float16* ap = abase + (dt * 16 + l15) * LDS_ROW;
                v8h alo = *(const v8h*)ap;
                v8h ahi = *(const v8h*)(ap + 8);
                v16h a = __builtin_shufflevector(alo, ahi,
                          0,1,2,3,4,5,6,7,8,9,10,11,12,13,14,15);
                acc[0][dt] = __builtin_amdgcn_wmma_f32_16x16x32_f16(
                    false, a, false, hfrag[0], (short)0, acc[0][dt], false, false);
                acc[1][dt] = __builtin_amdgcn_wmma_f32_16x16x32_f16(
                    false, a, false, hfrag[1], (short)0, acc[1][dt], false, false);
            }
        }

        if (s + 1 < NSTAGE) cvt_store(buf ^ 1);     // fill other buffer
        __syncthreads();
    }

    // ---- epilogue: each lane holds 8 consecutive d -> b128 stores
#pragma unroll
    for (int mt = 0; mt < 2; ++mt) {
        const int m = MB + mt * 16 + l15;
        float* op = out + ((size_t)m * NBRANCH + i) * ED + DB + lhi * 8;
#pragma unroll
        for (int dt = 0; dt < 8; ++dt) {
            v8f c = acc[mt][dt];
            v4f lo = {c[0], c[1], c[2], c[3]};
            v4f hi = {c[4], c[5], c[6], c[7]};
            *(v4f*)(op + dt * 16)     = lo;
            *(v4f*)(op + dt * 16 + 4) = hi;
        }
    }
}

extern "C" void kernel_launch(void* const* d_in, const int* in_sizes, int n_in,
                              void* d_out, int out_size, void* d_ws, size_t ws_size,
                              hipStream_t stream) {
    (void)in_sizes; (void)n_in; (void)d_ws; (void)ws_size; (void)out_size;
    const float* x  = (const float*)d_in[0];
    const float* W1 = (const float*)d_in[1];
    const float* b1 = (const float*)d_in[2];
    const float* W2 = (const float*)d_in[3];
    const float* b2 = (const float*)d_in[4];
    float* out = (float*)d_out;

    dim3 grid(ED / DBLK, B_TOT / MBLK, NBRANCH);   // (8, 8, 36) = 2304 blocks
    fused_mlp_wmma_kernel<<<grid, 256, 0, stream>>>(x, W1, b1, W2, b2, out);
}
